// SparseMoE_8100308320396
// MI455X (gfx1250) — compile-verified, matrix operand order checked by reading, not run on hardware
//
#include <hip/hip_runtime.h>

typedef __attribute__((ext_vector_type(16))) __bf16 v16bf;
typedef __attribute__((ext_vector_type(8)))  float  v8f;

union BFrag { unsigned int u[8]; v16bf v; };

#define BM 128
#define BN 128
#define BK 32
#define SPAD 34    // padded K-stride (bf16) for VGPR-staged tiles: 17-dword rows -> conflict-free
#define SPADA 40   // padded K-stride for async-staged tile: 80B rows (16B aligned), 20-dword stride

__device__ __forceinline__ unsigned short f2bf(float f) {
    unsigned int u = __float_as_uint(f);
    u += 0x7FFFu + ((u >> 16) & 1u);          // round-to-nearest-even
    return (unsigned short)(u >> 16);
}
__device__ __forceinline__ unsigned int pk2(unsigned short lo, unsigned short hi) {
    return (unsigned int)lo | ((unsigned int)hi << 16);
}

// Load a 16x32 bf16 fragment (A layout; B tiles are stored transposed so the
// same loader yields the B operand). ISA layout: lane holds row (lane&15),
// K-half kh = 8*(lane>=16); u[0..3] = K kh..kh+7, u[4..7] = K 16+kh..16+kh+7.
__device__ __forceinline__ void load_frag(const unsigned short* lds, int stride, int rowBase, int lane, BFrag& f) {
    const int m  = rowBase + (lane & 15);
    const int kh = (lane >> 4) << 3;
    const unsigned int* q = (const unsigned int*)lds;
    const int b = (m * stride + kh) >> 1;
    #pragma unroll
    for (int j = 0; j < 4; ++j) { f.u[j] = q[b + j]; f.u[4 + j] = q[b + 8 + j]; }
}

// ---------------- router: logits -> softmax -> top4 -> renorm; + shared sigmoid gate ----
__global__ __launch_bounds__(256) void moe_router(
    const float* __restrict__ X, const float* __restrict__ GW, const float* __restrict__ SGW,
    float* __restrict__ gates, float* __restrict__ sg, int N)
{
    const int wid  = threadIdx.x >> 5;
    const int lane = threadIdx.x & 31;
    const int tok  = blockIdx.x * 8 + wid;
    if (tok >= N) return;

    float acc[8] = {0.f,0.f,0.f,0.f,0.f,0.f,0.f,0.f};
    float accs = 0.f;
    const float* xr = X + (size_t)tok * 2048;
    for (int c = lane; c < 2048; c += 32) {
        const float xv = xr[c];
        const float* gw = GW + (size_t)c * 8;
        #pragma unroll
        for (int e = 0; e < 8; ++e) acc[e] += xv * gw[e];
        accs += xv * SGW[c];
    }
    #pragma unroll
    for (int e = 0; e < 8; ++e)
        for (int off = 16; off; off >>= 1) acc[e] += __shfl_xor(acc[e], off, 32);
    for (int off = 16; off; off >>= 1) accs += __shfl_xor(accs, off, 32);

    if (lane == 0) {
        float mx = acc[0];
        #pragma unroll
        for (int e = 1; e < 8; ++e) mx = fmaxf(mx, acc[e]);
        float p[8]; float s = 0.f;
        #pragma unroll
        for (int e = 0; e < 8; ++e) { p[e] = __expf(acc[e] - mx); s += p[e]; }
        const float inv_s = 1.f / s;
        #pragma unroll
        for (int e = 0; e < 8; ++e) p[e] *= inv_s;
        float out[8] = {0.f,0.f,0.f,0.f,0.f,0.f,0.f,0.f};
        float sel = 0.f;
        for (int k = 0; k < 4; ++k) {                  // top_k = 4
            int bj = 0; float bv = p[0];
            #pragma unroll
            for (int e = 1; e < 8; ++e) if (p[e] > bv) { bv = p[e]; bj = e; }
            out[bj] = bv; sel += bv; p[bj] = -1.f;
        }
        const float inv = 1.f / sel;
        #pragma unroll
        for (int e = 0; e < 8; ++e) gates[(size_t)tok * 8 + e] = out[e] * inv;
        sg[tok] = 1.f / (1.f + __expf(-accs));
    }
}

// ---------------- H = silu(X*Wg) .* (X*Wu), H stored bf16 [N][1408] ----------------
__global__ __launch_bounds__(256) void moe_gateup(
    const float* __restrict__ X, const float* __restrict__ Wg, const float* __restrict__ Wu,
    unsigned short* __restrict__ H, int Ktot, int ldw, int wcol0)
{
    __shared__ unsigned short Al [BM * SPAD];
    __shared__ unsigned short Bgl[BN * SPAD];
    __shared__ unsigned short Bul[BN * SPAD];

    const int tid  = threadIdx.x;
    const int lane = tid & 31;
    const int wid  = tid >> 5;
    const int wm   = wid & 1;          // 2 wave rows  -> 64 M-rows each
    const int wn   = wid >> 1;         // 4 wave cols  -> 32 N-cols each
    const int m0   = blockIdx.y * BM;
    const int i0   = blockIdx.x * BN;

    const v8f z = {0.f,0.f,0.f,0.f,0.f,0.f,0.f,0.f};
    v8f accg[4][2], accu[4][2];
    #pragma unroll
    for (int a = 0; a < 4; ++a)
        #pragma unroll
        for (int b = 0; b < 2; ++b) { accg[a][b] = z; accu[a][b] = z; }

    const int arow = tid >> 3, acol = (tid & 7) << 2;
    const int brow = tid >> 5, bcol = (tid & 31) << 2;

    for (int kk = 0; kk < Ktot; kk += BK) {
        // stage X tile (BM x BK) fp32 -> bf16 row-major
        #pragma unroll
        for (int it = 0; it < 4; ++it) {
            const int r = arow + it * 32;
            const float4 v = *(const float4*)(X + (size_t)(m0 + r) * Ktot + kk + acol);
            unsigned int* dst = (unsigned int*)&Al[r * SPAD + acol];
            dst[0] = pk2(f2bf(v.x), f2bf(v.y));
            dst[1] = pk2(f2bf(v.z), f2bf(v.w));
        }
        // stage Wg/Wu tiles (BK x BN) fp32 -> bf16, transposed to [BN][BK]
        #pragma unroll
        for (int it = 0; it < 4; ++it) {
            const int kr = brow + it * 8;
            const size_t off = (size_t)(kk + kr) * ldw + wcol0 + i0 + bcol;
            const float4 g = *(const float4*)(Wg + off);
            const float4 u = *(const float4*)(Wu + off);
            Bgl[(bcol + 0) * SPAD + kr] = f2bf(g.x);
            Bgl[(bcol + 1) * SPAD + kr] = f2bf(g.y);
            Bgl[(bcol + 2) * SPAD + kr] = f2bf(g.z);
            Bgl[(bcol + 3) * SPAD + kr] = f2bf(g.w);
            Bul[(bcol + 0) * SPAD + kr] = f2bf(u.x);
            Bul[(bcol + 1) * SPAD + kr] = f2bf(u.y);
            Bul[(bcol + 2) * SPAD + kr] = f2bf(u.z);
            Bul[(bcol + 3) * SPAD + kr] = f2bf(u.w);
        }
        if (kk + BK < Ktot) {   // lowers to global_prefetch_b8
            __builtin_prefetch(X  + (size_t)(m0 + arow) * Ktot + kk + BK + acol, 0, 1);
            __builtin_prefetch(Wg + (size_t)(kk + BK + brow) * ldw + wcol0 + i0 + bcol, 0, 1);
            __builtin_prefetch(Wu + (size_t)(kk + BK + brow) * ldw + wcol0 + i0 + bcol, 0, 1);
        }
        __syncthreads();

        BFrag af[4], bg[2], bu[2];
        #pragma unroll
        for (int tm = 0; tm < 4; ++tm) load_frag(Al, SPAD, wm * 64 + tm * 16, lane, af[tm]);
        #pragma unroll
        for (int tn = 0; tn < 2; ++tn) {
            load_frag(Bgl, SPAD, wn * 32 + tn * 16, lane, bg[tn]);
            load_frag(Bul, SPAD, wn * 32 + tn * 16, lane, bu[tn]);
        }
        #pragma unroll
        for (int tm = 0; tm < 4; ++tm)
            #pragma unroll
            for (int tn = 0; tn < 2; ++tn) {
                accg[tm][tn] = __builtin_amdgcn_wmma_f32_16x16x32_bf16(
                    false, af[tm].v, false, bg[tn].v, (short)0, accg[tm][tn], false, false);
                accu[tm][tn] = __builtin_amdgcn_wmma_f32_16x16x32_bf16(
                    false, af[tm].v, false, bu[tn].v, (short)0, accu[tm][tn], false, false);
            }
        __syncthreads();
    }

    // epilogue: H = silu(g) * u, bf16 store. C/D layout: M = r + 8*(lane>=16), N = lane&15.
    #pragma unroll
    for (int tm = 0; tm < 4; ++tm)
        #pragma unroll
        for (int tn = 0; tn < 2; ++tn) {
            const int n    = i0 + wn * 32 + tn * 16 + (lane & 15);
            const int mrow = m0 + wm * 64 + tm * 16 + ((lane >> 4) << 3);
            #pragma unroll
            for (int r = 0; r < 8; ++r) {
                const float g = accg[tm][tn][r];
                const float h = (g / (1.f + __expf(-g))) * accu[tm][tn][r];
                H[(size_t)(mrow + r) * 1408 + n] = f2bf(h);
            }
        }
}

// ---------------- Out (+)= gate .* (H * Wd) ----------------
__global__ __launch_bounds__(256) void moe_down(
    const unsigned short* __restrict__ H, const float* __restrict__ Wd,
    const float* __restrict__ gate, int gstride, float* __restrict__ Out, int beta)
{
    __shared__ unsigned short Al[BM * SPADA];   // async-staged, 16B-aligned rows
    __shared__ unsigned short Bl[BN * SPAD];

    const int Ktot = 1408, ldw = 2048;
    const int tid  = threadIdx.x;
    const int lane = tid & 31;
    const int wid  = tid >> 5;
    const int wm   = wid & 1;
    const int wn   = wid >> 1;
    const int m0   = blockIdx.y * BM;
    const int n0   = blockIdx.x * BN;

    const v8f z = {0.f,0.f,0.f,0.f,0.f,0.f,0.f,0.f};
    v8f acc[4][2];
    #pragma unroll
    for (int a = 0; a < 4; ++a)
        #pragma unroll
        for (int b = 0; b < 2; ++b) acc[a][b] = z;

    const int r4  = tid >> 2, seg = (tid & 3) << 3;   // A stage: 64 rows x 4 segs of 8 bf16 (16B)
    const int brow = tid >> 5, bcol = (tid & 31) << 2;
    const unsigned albase = (unsigned)(size_t)(void*)Al;   // LDS offset = low 32 bits

    for (int kk = 0; kk < Ktot; kk += BK) {
        // stage H tile (already bf16): async DMA global -> LDS, no VGPR round-trip
        #pragma unroll
        for (int it = 0; it < 2; ++it) {
            const int r = r4 + it * 64;
            const unsigned short* gsrc = H + (size_t)(m0 + r) * 1408 + kk + seg;
            const unsigned dls = albase + (unsigned)(r * SPADA + seg) * 2u;
            asm volatile("global_load_async_to_lds_b128 %0, %1, off"
                         :: "v"(dls), "v"(gsrc) : "memory");
        }
        // stage Wd tile (BK x BN) fp32 -> bf16 transposed
        #pragma unroll
        for (int it = 0; it < 4; ++it) {
            const int kr = brow + it * 8;
            const float4 w = *(const float4*)(Wd + (size_t)(kk + kr) * ldw + n0 + bcol);
            Bl[(bcol + 0) * SPAD + kr] = f2bf(w.x);
            Bl[(bcol + 1) * SPAD + kr] = f2bf(w.y);
            Bl[(bcol + 2) * SPAD + kr] = f2bf(w.z);
            Bl[(bcol + 3) * SPAD + kr] = f2bf(w.w);
        }
        if (kk + BK < Ktot)
            __builtin_prefetch(Wd + (size_t)(kk + BK + brow) * ldw + n0 + bcol, 0, 1);
        asm volatile("s_wait_asynccnt 0x0" ::: "memory");
        __syncthreads();

        BFrag af[4], bf[2];
        #pragma unroll
        for (int tm = 0; tm < 4; ++tm) load_frag(Al, SPADA, wm * 64 + tm * 16, lane, af[tm]);
        #pragma unroll
        for (int tn = 0; tn < 2; ++tn) load_frag(Bl, SPAD, wn * 32 + tn * 16, lane, bf[tn]);
        #pragma unroll
        for (int tm = 0; tm < 4; ++tm)
            #pragma unroll
            for (int tn = 0; tn < 2; ++tn)
                acc[tm][tn] = __builtin_amdgcn_wmma_f32_16x16x32_bf16(
                    false, af[tm].v, false, bf[tn].v, (short)0, acc[tm][tn], false, false);
        __syncthreads();
    }

    #pragma unroll
    for (int tm = 0; tm < 4; ++tm)
        #pragma unroll
        for (int tn = 0; tn < 2; ++tn) {
            const int n    = n0 + wn * 32 + tn * 16 + (lane & 15);
            const int mrow = m0 + wm * 64 + tm * 16 + ((lane >> 4) << 3);
            #pragma unroll
            for (int r = 0; r < 8; ++r) {
                const float gv  = gate[(size_t)(mrow + r) * gstride];
                const float val = gv * acc[tm][tn][r];
                const size_t o  = (size_t)(mrow + r) * 2048 + n;
                Out[o] = beta ? (Out[o] + val) : val;
            }
        }
}

extern "C" void kernel_launch(void* const* d_in, const int* in_sizes, int n_in,
                              void* d_out, int out_size, void* d_ws, size_t ws_size,
                              hipStream_t stream) {
    (void)in_sizes; (void)n_in; (void)out_size; (void)ws_size;
    const float* X   = (const float*)d_in[0];   // [8192, 2048]
    const float* GW  = (const float*)d_in[1];   // [2048, 8]
    const float* EG  = (const float*)d_in[2];   // [8, 2048, 1408]
    const float* EU  = (const float*)d_in[3];   // [8, 2048, 1408]
    const float* ED  = (const float*)d_in[4];   // [8, 1408, 2048]
    const float* SGw = (const float*)d_in[5];   // [2048, 5632]
    const float* SUw = (const float*)d_in[6];   // [2048, 5632]
    const float* SDw = (const float*)d_in[7];   // [5632, 2048]
    const float* SEG = (const float*)d_in[8];   // [2048, 1]
    float* Out = (float*)d_out;                  // [8192, 2048]

    const int N = 8192, C = 2048, I = 1408, E = 8;

    float* gates = (float*)d_ws;                         // [N, 8]
    float* sg    = gates + (size_t)N * E;                // [N]
    unsigned short* Hbuf = (unsigned short*)(sg + N);    // [N, 1408] bf16

    moe_router<<<N / 8, 256, 0, stream>>>(X, GW, SEG, gates, sg, N);

    dim3 blk(256);
    dim3 gGU(I / BN, N / BM);   // 11 x 64
    dim3 gDN(C / BN, N / BM);   // 16 x 64

    for (int e = 0; e < 8; ++e) {
        moe_gateup<<<gGU, blk, 0, stream>>>(X, EG + (size_t)e * C * I, EU + (size_t)e * C * I,
                                            Hbuf, C, I, 0);
        moe_down<<<gDN, blk, 0, stream>>>(Hbuf, ED + (size_t)e * I * C,
                                          gates + e, E, Out, e > 0);
    }
    // shared expert: 4 chunks of I=1408 columns/rows, gated by sigmoid, always accumulate
    for (int ch = 0; ch < 4; ++ch) {
        moe_gateup<<<gGU, blk, 0, stream>>>(X, SGw, SUw, Hbuf, C, 5632, ch * I);
        moe_down<<<gDN, blk, 0, stream>>>(Hbuf, SDw + (size_t)ch * I * C,
                                          sg, 1, Out, 1);
    }
}
